// multiHeadAttention_89249420411454
// MI455X (gfx1250) — matrix operand order repaired and checked by reading, not compiled, not run on hardware
//
#include <hip/hip_runtime.h>
#include <stdint.h>

#define EMB 384
#define HEADS 6
#define HSZ 64
#define BB 128
#define TT 256
#define TH 128          // key columns handled per block (split over blockIdx.z)
#define SLD 136         // padded leading dim of score tile buffer
#define VLD 136         // padded leading dim of transposed V
#define WLD 392         // padded leading dim of transposed staged weights (64 x 392)
#define KLD 40          // padded leading dim of transposed Wproj chunk (384 x 40)

typedef __bf16 v16bf __attribute__((ext_vector_type(16)));
typedef float  v8f   __attribute__((ext_vector_type(8)));

union V16 {
  v16bf v;
  unsigned short s[16];
  uint4 q[2];
};
union V8B {
  unsigned short s[8];
  uint4 q;
};

__device__ __forceinline__ unsigned short f2bf(float f) {
  union { __bf16 h; unsigned short u; } c;
  c.h = (__bf16)f;              // native v_cvt_pk_bf16_f32 when paired
  return c.u;
}
__device__ __forceinline__ float bf2f(unsigned short h) {
  union { unsigned int u; float f; } c; c.u = ((unsigned int)h) << 16;
  return c.f;
}

__device__ __forceinline__ v8f wmma_bf16(const V16& a, const V16& b, v8f c) {
  return __builtin_amdgcn_wmma_f32_16x16x32_bf16(false, a.v, false, b.v,
                                                 (short)0, c, false, false);
}

// A-tile 16x32 (bf16) from row-major [rows x ld] bf16 storage (LDS or global);
// two b128 loads per lane.  Lane L holds row M=L&15; elems i=0..7 -> K=8*half+i,
// i=8..15 -> K=16+8*half+(i-8).
__device__ __forceinline__ void load_a_bf16(const unsigned short* S, int row_base,
                                            int ld, int kc, int lane, V16& a) {
  const unsigned short* p = S + (size_t)(row_base + (lane & 15)) * ld + kc + 8 * (lane >> 4);
  a.q[0] = *(const uint4*)p;
  a.q[1] = *(const uint4*)(p + 16);
}

// A-tile 16x32 from row-major f32 global memory; 4x global_load_b128 + packed cvt.
__device__ __forceinline__ void load_a_f32g(const float* X, int row_base, int ld,
                                            int kc, int lane, V16& a) {
  const float* p = X + (size_t)(row_base + (lane & 15)) * ld + kc + 8 * (lane >> 4);
  float4 f0 = *(const float4*)(p);
  float4 f1 = *(const float4*)(p + 4);
  float4 f2 = *(const float4*)(p + 16);
  float4 f3 = *(const float4*)(p + 20);
  a.v[0] = (__bf16)f0.x;  a.v[1] = (__bf16)f0.y;  a.v[2]  = (__bf16)f0.z;  a.v[3]  = (__bf16)f0.w;
  a.v[4] = (__bf16)f1.x;  a.v[5] = (__bf16)f1.y;  a.v[6]  = (__bf16)f1.z;  a.v[7]  = (__bf16)f1.w;
  a.v[8] = (__bf16)f2.x;  a.v[9] = (__bf16)f2.y;  a.v[10] = (__bf16)f2.z;  a.v[11] = (__bf16)f2.w;
  a.v[12] = (__bf16)f3.x; a.v[13] = (__bf16)f3.y; a.v[14] = (__bf16)f3.z;  a.v[15] = (__bf16)f3.w;
}

// B-tile 32x16 where K is contiguous in memory (matrix stored [N x ld], K along row):
// B[k][n] = M[n_base + (lane&15)][k]; elems i=0..7 -> K=8*half+i, i=8..15 -> K=16+8*half+(i-8); two ds_load_b128 per lane.
__device__ __forceinline__ void load_b_contig(const unsigned short* S, int n_base,
                                              int ld, int kc, int lane, V16& b) {
  const unsigned short* p = S + (n_base + (lane & 15)) * ld + kc + 8 * (lane >> 4);
  b.q[0] = *(const uint4*)p;
  b.q[1] = *(const uint4*)(p + 16);
}

__global__ void __launch_bounds__(128) zero_kernel(float4* p, int n4) {
  int i = blockIdx.x * blockDim.x + threadIdx.x;
  if (i < n4) p[i] = make_float4(0.f, 0.f, 0.f, 0.f);
}

// One-time x f32 -> bf16 conversion (x is re-read 12x per batch row by attn blocks).
__global__ void __launch_bounds__(256) cvt_kernel(const float* __restrict__ x,
                                                  unsigned short* __restrict__ xb, int n8) {
  int i = blockIdx.x * blockDim.x + threadIdx.x;
  if (i >= n8) return;
  const float4 f0 = *(const float4*)(x + (size_t)i * 8);
  const float4 f1 = *(const float4*)(x + (size_t)i * 8 + 4);
  V8B o;
  o.s[0] = f2bf(f0.x); o.s[1] = f2bf(f0.y); o.s[2] = f2bf(f0.z); o.s[3] = f2bf(f0.w);
  o.s[4] = f2bf(f1.x); o.s[5] = f2bf(f1.y); o.s[6] = f2bf(f1.z); o.s[7] = f2bf(f1.w);
  *(uint4*)(xb + (size_t)i * 8) = o.q;
}

__global__ void __launch_bounds__(128)
attn_kernel(const unsigned short* __restrict__ xbf, const float* __restrict__ Wq,
            const float* __restrict__ Wk, const float* __restrict__ Wv,
            float* __restrict__ po) {
  __shared__ __align__(16) unsigned short sQ[TT * HSZ];        // 32 KB
  __shared__ __align__(16) unsigned short sK[TH * HSZ];        // 16 KB (this block's key rows)
  __shared__ __align__(16) unsigned short sVt[HSZ * VLD];      // 17 KB (V transposed: [d][s])
  __shared__ __align__(16) unsigned short sRegion[TT * SLD];   // 68 KB (weights stage, then S)
  __shared__ float colZ[TH];

  unsigned short* sWt = sRegion;   // 64 x WLD transposed weights (dead before S is written)
  unsigned short* sS  = sRegion;   // 256 x SLD exp'd masked scores

  const int b = blockIdx.x, h = blockIdx.y, kh = blockIdx.z;
  const int s0 = kh * TH;                                      // global key-column base
  const int tid = threadIdx.x, lane = tid & 31, wave = tid >> 5;
  const int radd = 8 * (lane >> 4), col0 = lane & 15;
  const unsigned short* xb = xbf + (size_t)b * TT * EMB;       // bf16 row-major [t][e]
  const float scale = 0.05103103630798288f;                    // 384^-0.5

  // ---- Q = x_b @ Wq[h]  (all 256 query rows) -> sQ row-major [t][d]
  {
    const float* W = Wq + (size_t)h * EMB * HSZ;
    for (int i = tid; i < EMB * HSZ; i += 128)
      sWt[(i & 63) * WLD + (i >> 6)] = f2bf(W[i]);             // stage transposed [c][k]
    __syncthreads();
#pragma unroll
    for (int si = 0; si < 4; ++si) {
      const int strip = (wave + 4 * si) * 16;
      v8f acc[4] = {};
      for (int kc = 0; kc < EMB; kc += 32) {
        V16 a; load_a_bf16(xb, strip, EMB, kc, lane, a);
        if (kc + 32 < EMB)
          __builtin_prefetch(xb + (size_t)(strip + col0) * EMB + kc + 32, 0, 1);
#pragma unroll
        for (int n = 0; n < 4; ++n) {
          V16 bm; load_b_contig(sWt, n * 16, WLD, kc, lane, bm);
          acc[n] = wmma_bf16(a, bm, acc[n]);
        }
      }
#pragma unroll
      for (int n = 0; n < 4; ++n)
#pragma unroll
        for (int r = 0; r < 8; ++r)
          sQ[(strip + r + radd) * HSZ + n * 16 + col0] = f2bf(acc[n][r]);
    }
    __syncthreads();
  }

  // ---- K rows [s0, s0+128) -> sK row-major [s_local][d]
  {
    const float* W = Wk + (size_t)h * EMB * HSZ;
    for (int i = tid; i < EMB * HSZ; i += 128)
      sWt[(i & 63) * WLD + (i >> 6)] = f2bf(W[i]);
    __syncthreads();
#pragma unroll
    for (int si = 0; si < 2; ++si) {
      const int strip = (wave + 4 * si) * 16;
      v8f acc[4] = {};
      for (int kc = 0; kc < EMB; kc += 32) {
        V16 a; load_a_bf16(xb, s0 + strip, EMB, kc, lane, a);
#pragma unroll
        for (int n = 0; n < 4; ++n) {
          V16 bm; load_b_contig(sWt, n * 16, WLD, kc, lane, bm);
          acc[n] = wmma_bf16(a, bm, acc[n]);
        }
      }
#pragma unroll
      for (int n = 0; n < 4; ++n)
#pragma unroll
        for (int r = 0; r < 8; ++r)
          sK[(strip + r + radd) * HSZ + n * 16 + col0] = f2bf(acc[n][r]);
    }
    __syncthreads();
  }

  // ---- V rows [s0, s0+128) -> sVt TRANSPOSED [d][s_local] (B-layout friendly)
  {
    const float* W = Wv + (size_t)h * EMB * HSZ;
    for (int i = tid; i < EMB * HSZ; i += 128)
      sWt[(i & 63) * WLD + (i >> 6)] = f2bf(W[i]);
    __syncthreads();
#pragma unroll
    for (int si = 0; si < 2; ++si) {
      const int strip = (wave + 4 * si) * 16;
      v8f acc[4] = {};
      for (int kc = 0; kc < EMB; kc += 32) {
        V16 a; load_a_bf16(xb, s0 + strip, EMB, kc, lane, a);
#pragma unroll
        for (int n = 0; n < 4; ++n) {
          V16 bm; load_b_contig(sWt, n * 16, WLD, kc, lane, bm);
          acc[n] = wmma_bf16(a, bm, acc[n]);
        }
      }
#pragma unroll
      for (int n = 0; n < 4; ++n)
#pragma unroll
        for (int r = 0; r < 8; ++r)
          sVt[(n * 16 + col0) * VLD + strip + r + radd] = f2bf(acc[n][r]);
    }
    __syncthreads();
  }

  // ---- Pass 1: sS[t][sL] = exp(scale * q_t . k_s) with causal mask (t >= s)
  // (reference softmaxes over the QUERY axis per key column)
  for (int si = 0; si < 4; ++si) {
    const int strip = (wave + 4 * si) * 16;
    V16 a0, a1;
    load_a_bf16(sQ, strip, HSZ, 0, lane, a0);
    load_a_bf16(sQ, strip, HSZ, 32, lane, a1);
#pragma unroll
    for (int nt = 0; nt < 8; ++nt) {
      v8f acc = {};
      V16 bm;
      load_b_contig(sK, nt * 16, HSZ, 0, lane, bm);
      acc = wmma_bf16(a0, bm, acc);
      load_b_contig(sK, nt * 16, HSZ, 32, lane, bm);
      acc = wmma_bf16(a1, bm, acc);
      const int sL = nt * 16 + col0, sG = s0 + sL;
#pragma unroll
      for (int r = 0; r < 8; ++r) {
        const int t = strip + r + radd;
        const float e = (t >= sG) ? __expf(acc[r] * scale) : 0.0f;
        sS[t * SLD + sL] = f2bf(e);
      }
    }
  }
  __syncthreads();

  // ---- Column sums (deterministic fixed-order sweep), fold 1/Z into V rows
  if (tid < TH) {
    float z = 0.0f;
    for (int t = 0; t < TT; ++t) z += bf2f(sS[t * SLD + tid]);
    colZ[tid] = 1.0f / z;
  }
  __syncthreads();
  for (int i = tid; i < HSZ * TH; i += 128) {
    const int d = i >> 7, sl = i & (TH - 1);
    const int idx = d * VLD + sl;
    sVt[idx] = f2bf(bf2f(sVt[idx]) * colZ[sl]);
  }
  __syncthreads();

  // ---- Pass 2: partial O = P~ @ V'; combine halves via 2-way global f32 atomics
  for (int si = 0; si < 4; ++si) {
    const int strip = (wave + 4 * si) * 16;
    v8f o[4] = {};
    for (int kc = 0; kc < TH; kc += 32) {
      V16 a; load_a_bf16(sS, strip, SLD, kc, lane, a);
#pragma unroll
      for (int n = 0; n < 4; ++n) {
        V16 bm; load_b_contig(sVt, n * 16, VLD, kc, lane, bm);
        o[n] = wmma_bf16(a, bm, o[n]);
      }
    }
#pragma unroll
    for (int n = 0; n < 4; ++n)
#pragma unroll
      for (int r = 0; r < 8; ++r) {
        const int t = strip + r + radd;
        atomicAdd(&po[((size_t)(b * TT + t)) * EMB + h * HSZ + n * 16 + col0], o[n][r]);
      }
  }
}

// Output projection: out(32768 x 384, f32) = po(f32) @ Wproj + bias.
// Wproj K-chunks (contiguous 48 KB) are DMA'd to LDS with async-to-LDS b128
// (ASYNCcnt-tracked, no VGPR round-trip), then converted/transposed to bf16.
__global__ void __launch_bounds__(128)
proj_kernel(const float* __restrict__ po, const float* __restrict__ Wp,
            const float* __restrict__ bias, float* __restrict__ out) {
  __shared__ __align__(16) float sF[EMB * 32];                 // 48 KB raw f32 chunk
  __shared__ __align__(16) unsigned short sWt[EMB * KLD];      // 30 KB, [c][k_local]
  const int tid = threadIdx.x, lane = tid & 31, wave = tid >> 5;
  const int radd = 8 * (lane >> 4), col0 = lane & 15;
  const int mrow = blockIdx.x * 64 + wave * 16;

  v8f acc[6][4] = {};
  for (int kc = 0; kc < EMB; kc += 32) {
    // async DMA: 3072 float4s, 24 per thread, straight into LDS
    const float* gbase = Wp + (size_t)kc * EMB;
#pragma unroll
    for (int j = 0; j < 24; ++j) {
      const int f4 = tid + j * 128;
      unsigned ldsa = (unsigned)(uintptr_t)&sF[f4 * 4];
      const float* g = gbase + f4 * 4;
      asm volatile("global_load_async_to_lds_b128 %0, %1, off"
                   :: "v"(ldsa), "v"(g) : "memory");
    }
    asm volatile("s_wait_asynccnt 0x0" ::: "memory");
    __syncthreads();
    // convert + transpose f32 chunk -> bf16 B panel [c][k]
    for (int i = tid; i < EMB * 32; i += 128) {
      const int k = i / EMB, c = i - k * EMB;
      sWt[c * KLD + k] = f2bf(sF[i]);
    }
    __syncthreads();
    V16 a; load_a_f32g(po, mrow, EMB, kc, lane, a);
#pragma unroll
    for (int g = 0; g < 6; ++g)
#pragma unroll
      for (int n = 0; n < 4; ++n) {
        V16 bm; load_b_contig(sWt, g * 64 + n * 16, KLD, 0, lane, bm);
        acc[g][n] = wmma_bf16(a, bm, acc[g][n]);
      }
    __syncthreads();
  }
#pragma unroll
  for (int g = 0; g < 6; ++g)
#pragma unroll
    for (int n = 0; n < 4; ++n) {
      const int c = g * 64 + n * 16 + col0;
      const float bv = bias[c];
#pragma unroll
      for (int r = 0; r < 8; ++r)
        out[(size_t)(mrow + r + radd) * EMB + c] = acc[g][n][r] + bv;
    }
}

extern "C" void kernel_launch(void* const* d_in, const int* in_sizes, int n_in,
                              void* d_out, int out_size, void* d_ws, size_t ws_size,
                              hipStream_t stream) {
  (void)in_sizes; (void)n_in; (void)out_size; (void)ws_size;
  const float* x  = (const float*)d_in[0];
  const float* Wq = (const float*)d_in[1];
  const float* Wk = (const float*)d_in[2];
  const float* Wv = (const float*)d_in[3];
  const float* Wp = (const float*)d_in[4];
  const float* bp = (const float*)d_in[5];

  const int nelem = BB * TT * EMB;                             // 12,582,912
  unsigned short* xbf = (unsigned short*)d_ws;                 // 25.2 MB bf16 x
  float* po = (float*)((char*)d_ws + (size_t)nelem * 2);       // 50.3 MB f32 accumulator

  const int n8 = nelem / 8, n4 = nelem / 4;
  cvt_kernel<<<(n8 + 255) / 256, 256, 0, stream>>>(x, xbf, n8);
  zero_kernel<<<(n4 + 127) / 128, 128, 0, stream>>>((float4*)po, n4);
  attn_kernel<<<dim3(BB, HEADS, 2), 128, 0, stream>>>(xbf, Wq, Wk, Wv, po);
  proj_kernel<<<(BB * TT) / 64, 128, 0, stream>>>(po, Wp, bp, (float*)d_out);
}
